// PillarEncoder_90649579749550
// MI455X (gfx1250) — matrix operand order repaired and checked by reading, not compiled, not run on hardware
//
#include <hip/hip_runtime.h>

// ---------------- CDNA5 WMMA vector types ----------------
typedef __attribute__((ext_vector_type(16))) _Float16 v16h;
typedef __attribute__((ext_vector_type(8)))  _Float16 h8;
typedef __attribute__((ext_vector_type(8)))  float    v8f;
typedef __attribute__((ext_vector_type(2)))  float    v2f;

#define B_      2
#define NPTS    250000
#define CIN     5
#define GRIDW   512
#define GRIDH   512
#define NCELL   (GRIDW * GRIDH)
#define COUT    128
#define X0_     (-51.2f)
#define VOX     (0.2f)
#define EPSBN   (1e-5f)

// LDS row strides (padded to dodge bank conflicts)
#define W1S 10   // floats per row of W1' [64 m][8 k]
#define W2S 72   // halfs per row of W2' [64 m][64 k]
#define W3S 72   // halfs per row of W3  [128 m][64 k]
#define HS  72   // halfs per row of hidden tile [16 n][64 k]
#define PS  12   // floats per row of paug tile  [16 n][8 k]

#define WPB 8    // waves per block
#define TPW 4    // tiles per wave
#define NTILES (B_ * NCELL / 16)   // 32768

// ---------------- kernel 1: winner init ----------------
__global__ void k_init(int* __restrict__ winner) {
  int i = blockIdx.x * blockDim.x + threadIdx.x;
  if (i < B_ * NCELL) winner[i] = -1;
}

// ---------------- kernel 2: last-write-wins = max point ordinal ----------------
__global__ void k_scatter(const float* __restrict__ pts, int* __restrict__ winner) {
  int i = blockIdx.x * blockDim.x + threadIdx.x;
  if (i >= B_ * NPTS) return;
  int b = i / NPTS, n = i - b * NPTS;
  const float* p = pts + ((size_t)b * NPTS + n) * CIN;
  float x = p[0], y = p[1];
  // trunc-toward-zero matches .astype(int32)
  int xi = (int)((x - X0_) / VOX);
  int yi = (int)((y - X0_) / VOX);
  if (xi >= 0 && xi < GRIDW && yi >= 0 && yi < GRIDH)
    atomicMax(&winner[b * NCELL + yi * GRIDW + xi], n);
}

// ---------------- kernel 3: WMMA MLP on winners + canvas write ----------------
__global__ __launch_bounds__(256) void k_mlp(
    const float* __restrict__ pts, const int* __restrict__ winner,
    const float* __restrict__ W1, const float* __restrict__ b1,
    const float* __restrict__ g1, const float* __restrict__ be1,
    const float* __restrict__ m1, const float* __restrict__ v1,
    const float* __restrict__ W2, const float* __restrict__ b2,
    const float* __restrict__ g2, const float* __restrict__ be2,
    const float* __restrict__ m2, const float* __restrict__ v2,
    const float* __restrict__ W3, const float* __restrict__ b3,
    float* __restrict__ out) {

  __shared__ __align__(16) float    s_w1[64 * W1S];     // W1*s1, [m][k]
  __shared__ __align__(16) _Float16 s_w2[64 * W2S];     // f16(W2*s2), [m][k]
  __shared__ __align__(16) _Float16 s_w3[COUT * W3S];   // f16(W3), [m][k]
  __shared__ __align__(16) float    s_b1[64], s_b2[64], s_b3[COUT];
  __shared__ __align__(16) float    s_pa[WPB][16 * PS]; // per-wave augmented inputs [n][k]
  __shared__ __align__(16) _Float16 s_h [WPB][16 * HS]; // per-wave hidden tile [n][k]
  __shared__ __align__(16) float    s_mk[WPB][16];      // per-cell validity mask

  const int tid = threadIdx.x;

  // ---- stage BN-folded weights into LDS (once per block) ----
  if (tid < 64) {
    int m = tid;
    float s1 = g1[m] * rsqrtf(v1[m] + EPSBN);
    s_b1[m] = (b1[m] - m1[m]) * s1 + be1[m];
    #pragma unroll
    for (int k = 0; k < 8; ++k) s_w1[m * W1S + k] = W1[k * 64 + m] * s1;
    float s2v = g2[m] * rsqrtf(v2[m] + EPSBN);
    s_b2[m] = (b2[m] - m2[m]) * s2v + be2[m];
  }
  if (tid < COUT) s_b3[tid] = b3[tid];
  for (int e = tid; e < 64 * 64; e += 256) {
    int m = e >> 6, k = e & 63;
    float s2v = g2[m] * rsqrtf(v2[m] + EPSBN);
    s_w2[m * W2S + k] = (_Float16)(W2[k * 64 + m] * s2v);
  }
  for (int e = tid; e < COUT * 64; e += 256) {
    int m = e >> 6, k = e & 63;
    s_w3[m * W3S + k] = (_Float16)(W3[k * COUT + m]);
  }
  __syncthreads();

  const int lane = tid & 31;
  const int wv   = tid >> 5;
  const int half = lane >> 4;   // l/16  (wave32)
  const int ln   = lane & 15;   // l%16

  float*    pa = &s_pa[wv][0];
  _Float16* hb = &s_h[wv][0];

  for (int t = 0; t < TPW; ++t) {
    int tile = (blockIdx.x * WPB + wv) * TPW + t;       // < NTILES by construction
    int b  = tile / (NCELL / 16);
    int cb = (tile - b * (NCELL / 16)) * 16;            // base cell of this 16-cell tile

    // ---- gather winner point, build 8 augmented features (lanes 0..15) ----
    if (half == 0) {
      int cell = cb + ln;
      int w = winner[b * NCELL + cell];
      float f0 = 0.f, f1 = 0.f, f2 = 0.f, f3 = 0.f, f4 = 0.f, f5 = 0.f, f6 = 0.f;
      if (w >= 0) {
        const float* p = pts + ((size_t)b * NPTS + w) * CIN;
        f0 = p[0]; f1 = p[1]; f2 = p[2]; f3 = p[3]; f4 = p[4];
        int xi = cell & (GRIDW - 1), yi = cell >> 9;
        f5 = f0 - (xi * VOX + (X0_ + 0.5f * VOX));
        f6 = f1 - (yi * VOX + (X0_ + 0.5f * VOX));
      }
      pa[ln * PS + 0] = f0; pa[ln * PS + 1] = f1; pa[ln * PS + 2] = f2;
      pa[ln * PS + 3] = f3; pa[ln * PS + 4] = f4; pa[ln * PS + 5] = f5;
      pa[ln * PS + 6] = f6; pa[ln * PS + 7] = 0.f;
      s_mk[wv][ln] = (w >= 0) ? 1.0f : 0.0f;
    }
    asm volatile("s_wait_dscnt 0" ::: "memory");  // wave-internal LDS visibility fence
    __builtin_amdgcn_wave_barrier();

    // ---- layer 1: H1^T = relu(W1'^T[64x8] @ Paug^T[8x16]), fp32 WMMA 16x16x4 ----
    // B frag (4x16 f32): elem e <-> B[4*ks + 2*half + e][ln]
    v2f bf0, bf1;
    bf0.x = pa[ln * PS + 2 * half + 0];
    bf0.y = pa[ln * PS + 2 * half + 1];
    bf1.x = pa[ln * PS + 4 + 2 * half + 0];
    bf1.y = pa[ln * PS + 4 + 2 * half + 1];

    #pragma unroll
    for (int mt = 0; mt < 4; ++mt) {
      v8f acc;
      #pragma unroll
      for (int r = 0; r < 8; ++r) acc[r] = s_b1[16 * mt + 8 * half + r];
      // A frag (16x4 f32): elem e <-> A[ln][4*ks + 2*half + e] = W1'[k][m]
      v2f a0, a1;
      a0.x = s_w1[(16 * mt + ln) * W1S + 2 * half + 0];
      a0.y = s_w1[(16 * mt + ln) * W1S + 2 * half + 1];
      a1.x = s_w1[(16 * mt + ln) * W1S + 4 + 2 * half + 0];
      a1.y = s_w1[(16 * mt + ln) * W1S + 4 + 2 * half + 1];
      acc = __builtin_amdgcn_wmma_f32_16x16x4_f32(false, a0, false, bf0, (short)0, acc, false, false);
      acc = __builtin_amdgcn_wmma_f32_16x16x4_f32(false, a1, false, bf1, (short)0, acc, false, false);
      // relu -> f16, D layout: reg r holds row m = 16*mt + 8*half + r, col n = ln
      h8 hv;
      #pragma unroll
      for (int r = 0; r < 8; ++r) hv[r] = (_Float16)fmaxf(acc[r], 0.f);
      *(h8*)(&hb[ln * HS + 16 * mt + 8 * half]) = hv;
    }

    // ---- layer 2: H2^T = relu(W2'^T[64x64] @ H1^T[64x16]), f16 WMMA 16x16x32 ----
    // B frag (32x16 f16): j<8 -> k = 32*s + 8*half + j ; j>=8 -> k = 32*s + 16 + 8*half + (j-8)
    v16h b20, b21;
    #pragma unroll
    for (int j = 0; j < 16; ++j) b20[j] = hb[ln * HS + 8 * half + (j & 7) + ((j >> 3) << 4)];
    #pragma unroll
    for (int j = 0; j < 16; ++j) b21[j] = hb[ln * HS + 32 + 8 * half + (j & 7) + ((j >> 3) << 4)];

    #pragma unroll
    for (int mt = 0; mt < 4; ++mt) {
      v8f acc;
      #pragma unroll
      for (int r = 0; r < 8; ++r) acc[r] = s_b2[16 * mt + 8 * half + r];
      // A frag (16x32 f16): j<8 -> k = 8*half+j ; j>=8 -> k = 16 + 8*half + (j-8)
      v16h a0, a1;
      #pragma unroll
      for (int j = 0; j < 8; ++j) {
        a0[j]     = s_w2[(16 * mt + ln) * W2S +      8 * half + j];
        a0[j + 8] = s_w2[(16 * mt + ln) * W2S + 16 + 8 * half + j];
        a1[j]     = s_w2[(16 * mt + ln) * W2S + 32 +      8 * half + j];
        a1[j + 8] = s_w2[(16 * mt + ln) * W2S + 32 + 16 + 8 * half + j];
      }
      acc = __builtin_amdgcn_wmma_f32_16x16x32_f16(false, a0, false, b20, (short)0, acc, false, false);
      acc = __builtin_amdgcn_wmma_f32_16x16x32_f16(false, a1, false, b21, (short)0, acc, false, false);
      h8 hv;
      #pragma unroll
      for (int r = 0; r < 8; ++r) hv[r] = (_Float16)fmaxf(acc[r], 0.f);
      *(h8*)(&hb[ln * HS + 16 * mt + 8 * half]) = hv;  // overwrite hidden tile (b2* already in regs)
    }

    // ---- layer 3: F^T = W3^T[128x64] @ H2^T[64x16], masked canvas write ----
    v16h b30, b31;
    #pragma unroll
    for (int j = 0; j < 16; ++j) b30[j] = hb[ln * HS + 8 * half + (j & 7) + ((j >> 3) << 4)];
    #pragma unroll
    for (int j = 0; j < 16; ++j) b31[j] = hb[ln * HS + 32 + 8 * half + (j & 7) + ((j >> 3) << 4)];

    float mk = s_mk[wv][ln];
    size_t obase = (size_t)b * COUT * NCELL + (size_t)cb + ln;

    #pragma unroll
    for (int mt = 0; mt < 8; ++mt) {
      v8f acc;
      #pragma unroll
      for (int r = 0; r < 8; ++r) acc[r] = s_b3[16 * mt + 8 * half + r];
      v16h a0, a1;
      #pragma unroll
      for (int j = 0; j < 8; ++j) {
        a0[j]     = s_w3[(16 * mt + ln) * W3S +      8 * half + j];
        a0[j + 8] = s_w3[(16 * mt + ln) * W3S + 16 + 8 * half + j];
        a1[j]     = s_w3[(16 * mt + ln) * W3S + 32 +      8 * half + j];
        a1[j + 8] = s_w3[(16 * mt + ln) * W3S + 32 + 16 + 8 * half + j];
      }
      acc = __builtin_amdgcn_wmma_f32_16x16x32_f16(false, a0, false, b30, (short)0, acc, false, false);
      acc = __builtin_amdgcn_wmma_f32_16x16x32_f16(false, a1, false, b31, (short)0, acc, false, false);
      // D layout: reg r = channel 16*mt + 8*half + r, lane%16 = cell -> 64B coalesced segments
      #pragma unroll
      for (int r = 0; r < 8; ++r)
        out[obase + (size_t)(16 * mt + 8 * half + r) * NCELL] = acc[r] * mk;
    }
  }
}

// ---------------- host launcher ----------------
extern "C" void kernel_launch(void* const* d_in, const int* in_sizes, int n_in,
                              void* d_out, int out_size, void* d_ws, size_t ws_size,
                              hipStream_t stream) {
  const float* pts = (const float*)d_in[0];
  const float* W1  = (const float*)d_in[1];
  const float* b1  = (const float*)d_in[2];
  const float* g1  = (const float*)d_in[3];
  const float* be1 = (const float*)d_in[4];
  const float* m1  = (const float*)d_in[5];
  const float* v1  = (const float*)d_in[6];
  const float* W2  = (const float*)d_in[7];
  const float* b2  = (const float*)d_in[8];
  const float* g2  = (const float*)d_in[9];
  const float* be2 = (const float*)d_in[10];
  const float* m2  = (const float*)d_in[11];
  const float* v2  = (const float*)d_in[12];
  const float* W3  = (const float*)d_in[13];
  const float* b3  = (const float*)d_in[14];
  float* out  = (float*)d_out;
  int*   winr = (int*)d_ws;   // 2 * 262144 ints = 2 MB scratch

  k_init<<<(B_ * NCELL + 255) / 256, 256, 0, stream>>>(winr);
  k_scatter<<<(B_ * NPTS + 255) / 256, 256, 0, stream>>>(pts, winr);
  k_mlp<<<NTILES / (WPB * TPW), 256, 0, stream>>>(
      pts, winr, W1, b1, g1, be1, m1, v1, W2, b2, g2, be2, m2, v2, W3, b3, out);
}
